// MultiHeadSelfAttention_23570780521210
// MI455X (gfx1250) — matrix operand order repaired and checked by reading, not compiled, not run on hardware
//
#include <hip/hip_runtime.h>
#include <hip/hip_bf16.h>

// ---------------------------------------------------------------------------
// MHSA: x[4,2048,1024] -> QKV proj -> 16-head attention (hd=64) -> out proj
//       -> LayerNorm.  All matmuls in bf16 WMMA (fp32 accumulate).
// A/K tiles staged via Tensor Data Mover; B/V tiles stored LDS-swizzled in
// WMMA fragment order so every lane loads its fragment as 2x ds_load_b128.
// ---------------------------------------------------------------------------

#define BDIM 4
#define SDIM 2048
#define DDIM 1024
#define HDIM 16
#define HD   64
#define MROWS (BDIM * SDIM)   // 8192

typedef __attribute__((ext_vector_type(16))) __bf16 bf16x16;
typedef __attribute__((ext_vector_type(8)))  __bf16 bf16x8;
typedef __attribute__((ext_vector_type(8)))  float  v8f;
typedef __attribute__((ext_vector_type(4)))  unsigned int u32x4;
typedef __attribute__((ext_vector_type(8)))  int i32x8;
typedef __attribute__((ext_vector_type(4)))  int i32x4;

#if defined(__has_builtin)
#  if __has_builtin(__builtin_amdgcn_tensor_load_to_lds)
#    define HAS_TDM 1
#  else
#    define HAS_TDM 0
#  endif
#  if __has_builtin(__builtin_amdgcn_s_wait_tensorcnt)
#    define WAIT_TENSORCNT0() __builtin_amdgcn_s_wait_tensorcnt(0)
#  else
#    define WAIT_TENSORCNT0() asm volatile("s_wait_tensorcnt 0" ::: "memory")
#  endif
#else
#  define HAS_TDM 0
#  define WAIT_TENSORCNT0()
#endif

__device__ __forceinline__ v8f wmma_bf16(bf16x16 a, bf16x16 b, v8f c) {
  // D = A(16x32) * B(32x16) + C, fp32 accumulate
  return __builtin_amdgcn_wmma_f32_16x16x32_bf16(
      /*neg_a=*/false, a, /*neg_b=*/false, b,
      /*c_mod=*/(short)0, c, /*reuse_a=*/false, /*reuse_b=*/false);
}

__device__ __forceinline__ unsigned lds_byte_off(const void* p) {
  return (unsigned)(size_t)p;
}

// 2D TDM tile load: tile_d0 = contiguous elements per row, tile_d1 = rows,
// stride0 = row stride in elements.  data_size = 2 bytes (bf16).
__device__ __forceinline__ void tdm_load_2d(unsigned lds_off, const void* gptr,
                                            unsigned tile_d0, unsigned tile_d1,
                                            unsigned long long stride0) {
#if HAS_TDM
  unsigned long long ga = (unsigned long long)(size_t)gptr;
  u32x4 g0;
  g0[0] = 1u;                                            // count=1 (user D#)
  g0[1] = lds_off;                                       // lds_addr
  g0[2] = (unsigned)(ga & 0xFFFFFFFFull);                // global_addr lo
  g0[3] = (unsigned)((ga >> 32) & 0x1FFFFFFull) | (2u << 30);  // hi | type=2
  i32x8 g1;
  g1[0] = (int)(1u << 16);                               // data_size=1 -> 2B
  g1[1] = (int)((tile_d0 & 0xFFFFu) << 16);              // tensor_dim0 lo16
  g1[2] = (int)(((tile_d0 >> 16) & 0xFFFFu) | ((tile_d1 & 0xFFFFu) << 16));
  g1[3] = (int)(((tile_d1 >> 16) & 0xFFFFu) | ((tile_d0 & 0xFFFFu) << 16)); // tile_dim0
  g1[4] = (int)(tile_d1 & 0xFFFFu);                      // tile_dim1, tile_dim2=0
  g1[5] = (int)(unsigned)(stride0 & 0xFFFFFFFFull);      // tensor_dim0_stride lo
  g1[6] = (int)(unsigned)((stride0 >> 32) & 0xFFFFull);  // stride hi16
  g1[7] = 0;
  i32x4 z4; z4[0] = 0; z4[1] = 0; z4[2] = 0; z4[3] = 0;
  i32x8 z8;
#pragma unroll
  for (int i = 0; i < 8; ++i) z8[i] = 0;
  // amdgpu-toolchain (clang-23) 6-arg form
  __builtin_amdgcn_tensor_load_to_lds(g0, g1, z4, z4, z8, 0);
#else
  (void)lds_off; (void)gptr; (void)tile_d0; (void)tile_d1; (void)stride0;
#endif
}

// ---- WMMA fragment loaders (contiguous k order: k = 16*(lane>>4) + j, wave32) ----------

__device__ __forceinline__ bf16x16 concat8(bf16x8 lo, bf16x8 hi) {
  return __builtin_shufflevector(lo, hi, 0, 1, 2, 3, 4, 5, 6, 7,
                                 8, 9, 10, 11, 12, 13, 14, 15);
}

// A 16x32 bf16 from row-major LDS: lane m = lane&15, kb = 16*(lane>>4);
// element j: K = kb+j.  Two aligned 16B runs.
__device__ __forceinline__ bf16x16 load_a_frag(const __bf16* base, int ld) {
  const int lane = threadIdx.x & 31;
  const int m = lane & 15;
  const int kb = (lane >> 4) * 16;
  const __bf16* p0 = base + m * ld + kb;
  bf16x8 lo = *(const bf16x8*)p0;
  bf16x8 hi = *(const bf16x8*)(p0 + 8);
  return concat8(lo, hi);
}

// B frag from K stored [key][hd]: K^T needed -> element j = hd (contiguous),
// lane n = key.  One aligned 32B run per lane.
__device__ __forceinline__ bf16x16 load_bT_frag(const __bf16* base, int ld) {
  const int lane = threadIdx.x & 31;
  const int n = lane & 15;
  const int kb = (lane >> 4) * 16;
  return *(const bf16x16*)(base + n * ld + kb);
}

// ---- shared 128x128 GEMM mainloop (K=1024, BK=32, 8 waves: 4M x 2N) -------
// Bs is stored swizzled in fragment order:
//   Bsw[(((nt*2+hh)*16+n)*16)+j] = B[hh*16+j][nt*16+n],  nt=0..7, hh=0..1
// so a lane's 16 fragment elements are one contiguous 32B run.

__device__ __forceinline__ void gemm_128x128_bf16(
    const __bf16* __restrict__ A,   // at (rowBase, 0), ld = 1024
    const __bf16* __restrict__ Bm,  // at (0, colBase), ld = 1024
    __bf16* As, __bf16* Bsw, v8f (&acc)[2][4]) {
  const int tid = threadIdx.x;
  const int wave = tid >> 5;
  const int wm = wave >> 1;
  const int wn = wave & 1;
  const int lane = tid & 31;
  const int n16 = lane & 15;
  const int hh = lane >> 4;
  for (int kk = 0; kk < 1024; kk += 32) {
#if HAS_TDM
    if (tid < 32) {
      if (tid == 0)
        tdm_load_2d(lds_byte_off(As), A + kk, 32u, 128u, 1024ull);  // in flight
    }
#else
#pragma unroll 4
    for (int i = tid; i < 128 * 32; i += 256) {
      int r = i >> 5, c = i & 31;
      As[i] = A[(size_t)r * 1024 + kk + c];
    }
#endif
    // swizzled B fill (overlaps the TDM transfer): 1024 4-element chunks
#pragma unroll
    for (int it = 0; it < 4; ++it) {
      const int i4 = tid + it * 256;
      const int k = i4 >> 5;            // 0..31
      const int c0 = (i4 & 31) * 4;     // 0..124
      const __bf16* gp = Bm + (size_t)(kk + k) * 1024 + c0;
      const int bh = k >> 4, bj = k & 15;
#pragma unroll
      for (int e = 0; e < 4; ++e) {
        const int c = c0 + e, nt = c >> 4, n = c & 15;
        Bsw[(((nt * 2 + bh) * 16 + n) << 4) + bj] = gp[e];
      }
    }
#if HAS_TDM
    if (tid < 32) WAIT_TENSORCNT0();
#endif
    __syncthreads();
    bf16x16 af0 = load_a_frag(As + (wm * 32) * 32, 32);
    bf16x16 af1 = load_a_frag(As + (wm * 32 + 16) * 32, 32);
#pragma unroll
    for (int nt = 0; nt < 4; ++nt) {
      const int ntg = wn * 4 + nt;
      bf16x16 bf = *(const bf16x16*)(Bsw + (((ntg * 2 + hh) * 16 + n16) << 4));
      acc[0][nt] = wmma_bf16(af0, bf, acc[0][nt]);
      acc[1][nt] = wmma_bf16(af1, bf, acc[1][nt]);
    }
    __syncthreads();
  }
}

// ---- kernels --------------------------------------------------------------

__global__ void cvt_f32_bf16(const float* __restrict__ s, __bf16* __restrict__ d, int n) {
  int i = blockIdx.x * blockDim.x + threadIdx.x;
  if (i < n) d[i] = (__bf16)s[i];
}

// QKV projection: z in {0,1,2} selects {q,k,v}.  Output bf16 in [B,H,S,hd].
__global__ __launch_bounds__(256) void qkv_gemm_kernel(
    const __bf16* __restrict__ xb,
    const __bf16* __restrict__ wqb, const __bf16* __restrict__ wkb,
    const __bf16* __restrict__ wvb,
    const float* __restrict__ bq, const float* __restrict__ bk,
    const float* __restrict__ bv,
    __bf16* __restrict__ q, __bf16* __restrict__ k, __bf16* __restrict__ v) {
  __shared__ alignas(32) __bf16 As[128 * 32];
  __shared__ alignas(32) __bf16 Bs[32 * 128];
  const int z = blockIdx.z;
  const __bf16* W = (z == 0) ? wqb : (z == 1) ? wkb : wvb;
  const float* bias = (z == 0) ? bq : (z == 1) ? bk : bv;
  __bf16* out = (z == 0) ? q : (z == 1) ? k : v;
  const int rowBase = blockIdx.x * 128;
  const int colBase = blockIdx.y * 128;
  v8f acc[2][4];
  const v8f zero = {0.f, 0.f, 0.f, 0.f, 0.f, 0.f, 0.f, 0.f};
#pragma unroll
  for (int mt = 0; mt < 2; ++mt)
#pragma unroll
    for (int nt = 0; nt < 4; ++nt) acc[mt][nt] = zero;

  gemm_128x128_bf16(xb + (size_t)rowBase * 1024, W + colBase, As, Bs, acc);

  const int tid = threadIdx.x, lane = tid & 31, wave = tid >> 5;
  const int wm = wave >> 1, wn = wave & 1;
  const int n16 = lane & 15, half = lane >> 4;
#pragma unroll
  for (int mt = 0; mt < 2; ++mt)
#pragma unroll
    for (int nt = 0; nt < 4; ++nt) {
      const int col = colBase + wn * 64 + nt * 16 + n16;
      const float bsv = bias[col];
      const int h = col >> 6, dd = col & 63;
#pragma unroll
      for (int r = 0; r < 8; ++r) {
        const int row = rowBase + wm * 32 + mt * 16 + half * 8 + r;  // b*S+s
        const int bb = row >> 11, ss = row & 2047;
        out[((((size_t)bb * HDIM + h) * SDIM + ss) << 6) + dd] =
            (__bf16)(acc[mt][nt][r] + bsv);
      }
    }
}

// Flash attention: block = 4 waves, each wave owns 16 q-rows for one (b,h).
// Vs stored swizzled: Vsw[((((t2*4+nt)*2+hh)*16+n)*16)+j] = V[t2*32+hh*16+j][nt*16+n]
__global__ __launch_bounds__(128) void attn_kernel(
    const __bf16* __restrict__ Q, const __bf16* __restrict__ K,
    const __bf16* __restrict__ V, __bf16* __restrict__ O) {
  __shared__ alignas(32) __bf16 Ks[64 * 64];
  __shared__ alignas(32) __bf16 Vs[64 * 64];
  __shared__ alignas(32) __bf16 Ps[4 * 16 * 64];
  const int tid = threadIdx.x;
  const int lane = tid & 31;
  const int wave = tid >> 5;
  const int n16 = lane & 15;
  const int half = lane >> 4;
  const int h = blockIdx.y, b = blockIdx.z;
  const size_t bh = ((size_t)b * HDIM + h) * SDIM;
  const int qBase = blockIdx.x * 64 + wave * 16;

  // resident Q fragments (A layout), hd = 64 -> two 16x32 frags
  bf16x16 aq[2];
  {
    const __bf16* qp = Q + (bh + qBase) * HD;
    const int kb = half * 16;
#pragma unroll
    for (int f = 0; f < 2; ++f) {
      const __bf16* p0 = qp + n16 * HD + f * 32 + kb;
      aq[f] = concat8(*(const bf16x8*)p0, *(const bf16x8*)(p0 + 8));
    }
  }

  float rmax[8], rsum[8];
  v8f oacc[4];
  const v8f zero = {0.f, 0.f, 0.f, 0.f, 0.f, 0.f, 0.f, 0.f};
#pragma unroll
  for (int r = 0; r < 8; ++r) { rmax[r] = -1e30f; rsum[r] = 0.f; }
#pragma unroll
  for (int nt = 0; nt < 4; ++nt) oacc[nt] = zero;

  const __bf16* kg = K + bh * HD;
  const __bf16* vg = V + bh * HD;

  for (int s0 = 0; s0 < SDIM; s0 += 64) {
#if HAS_TDM
    if (tid < 32) {
      if (tid == 0)
        tdm_load_2d(lds_byte_off(Ks), kg + (size_t)s0 * HD, 64u, 64u, 64ull);
    }
#else
#pragma unroll 4
    for (int i = tid; i < 64 * 64; i += 128) Ks[i] = kg[(size_t)s0 * HD + i];
#endif
    // swizzled V fill (overlaps K TDM): 1024 4-element chunks, 128 threads
#pragma unroll
    for (int it = 0; it < 8; ++it) {
      const int i4 = tid + it * 128;
      const int key = i4 >> 4;           // 0..63
      const int h0 = (i4 & 15) * 4;      // 0..60
      const __bf16* gp = vg + (size_t)(s0 + key) * HD + h0;
      const int t2 = key >> 5, vh = (key >> 4) & 1, vj = key & 15;
#pragma unroll
      for (int e = 0; e < 4; ++e) {
        const int hd = h0 + e, nt = hd >> 4, n = hd & 15;
        Vs[(((((t2 * 4 + nt) * 2 + vh) * 16) + n) << 4) + vj] = gp[e];
      }
    }
#if HAS_TDM
    if (tid < 32) WAIT_TENSORCNT0();
#endif
    __syncthreads();

    // scores: 4 tiles of 16q x 16k, accumulated over hd=64 (2 WMMAs each)
    v8f sc[4];
#pragma unroll
    for (int t = 0; t < 4; ++t) {
      v8f c = zero;
      c = wmma_bf16(aq[0], load_bT_frag(Ks + (t * 16) * HD, HD), c);
      c = wmma_bf16(aq[1], load_bT_frag(Ks + (t * 16) * HD + 32, HD), c);
      sc[t] = c;
    }
#pragma unroll
    for (int t = 0; t < 4; ++t)
#pragma unroll
      for (int r = 0; r < 8; ++r) sc[t][r] *= 0.125f;  // 1/sqrt(64)

    // streaming softmax
    float mnew[8], scl[8];
#pragma unroll
    for (int r = 0; r < 8; ++r) {
      float vmax = fmaxf(fmaxf(sc[0][r], sc[1][r]), fmaxf(sc[2][r], sc[3][r]));
#pragma unroll
      for (int off = 1; off < 16; off <<= 1) vmax = fmaxf(vmax, __shfl_xor(vmax, off));
      mnew[r] = fmaxf(rmax[r], vmax);
      scl[r] = __expf(rmax[r] - mnew[r]);
      rmax[r] = mnew[r];
    }
    float rs[8];
#pragma unroll
    for (int r = 0; r < 8; ++r) rs[r] = 0.f;
#pragma unroll
    for (int t = 0; t < 4; ++t)
#pragma unroll
      for (int r = 0; r < 8; ++r) {
        float p = __expf(sc[t][r] - mnew[r]);
        sc[t][r] = p;
        rs[r] += p;
      }
#pragma unroll
    for (int r = 0; r < 8; ++r) {
      float vsum = rs[r];
#pragma unroll
      for (int off = 1; off < 16; off <<= 1) vsum += __shfl_xor(vsum, off);
      rsum[r] = rsum[r] * scl[r] + vsum;
    }
#pragma unroll
    for (int nt = 0; nt < 4; ++nt)
#pragma unroll
      for (int r = 0; r < 8; ++r) oacc[nt][r] *= scl[r];

    // P (C/D layout) -> LDS -> A-frag layout, then P @ V
    __bf16* Pw = Ps + wave * 16 * 64;
#pragma unroll
    for (int t = 0; t < 4; ++t)
#pragma unroll
      for (int r = 0; r < 8; ++r)
        Pw[(r + half * 8) * 64 + t * 16 + n16] = (__bf16)sc[t][r];
    // same-wave LDS is in-order; compiler inserts dscnt waits before reads

#pragma unroll
    for (int t2 = 0; t2 < 2; ++t2) {
      bf16x16 ap = load_a_frag(Pw + t2 * 32, 64);
#pragma unroll
      for (int nt = 0; nt < 4; ++nt) {
        bf16x16 bv = *(const bf16x16*)(
            Vs + (((((t2 * 4 + nt) * 2 + half) * 16) + n16) << 4));
        oacc[nt] = wmma_bf16(ap, bv, oacc[nt]);
      }
    }
    __syncthreads();
  }

  // normalize and store to [B,S,ATTN_D] bf16
  const size_t orow0 = (size_t)b * SDIM + qBase;
#pragma unroll
  for (int nt = 0; nt < 4; ++nt)
#pragma unroll
    for (int r = 0; r < 8; ++r) {
      const int m = r + half * 8;
      const float val = oacc[nt][r] / rsum[r];
      O[(orow0 + m) * DDIM + h * HD + nt * 16 + n16] = (__bf16)val;
    }
}

// Output projection: attn[8192x1024] @ wo[1024x1024] + bo -> fp32
__global__ __launch_bounds__(256) void oproj_gemm_kernel(
    const __bf16* __restrict__ attnb, const __bf16* __restrict__ wob,
    const float* __restrict__ bo, float* __restrict__ proj) {
  __shared__ alignas(32) __bf16 As[128 * 32];
  __shared__ alignas(32) __bf16 Bs[32 * 128];
  const int rowBase = blockIdx.x * 128;
  const int colBase = blockIdx.y * 128;
  v8f acc[2][4];
  const v8f zero = {0.f, 0.f, 0.f, 0.f, 0.f, 0.f, 0.f, 0.f};
#pragma unroll
  for (int mt = 0; mt < 2; ++mt)
#pragma unroll
    for (int nt = 0; nt < 4; ++nt) acc[mt][nt] = zero;

  gemm_128x128_bf16(attnb + (size_t)rowBase * 1024, wob + colBase, As, Bs, acc);

  const int tid = threadIdx.x, lane = tid & 31, wave = tid >> 5;
  const int wm = wave >> 1, wn = wave & 1;
  const int n16 = lane & 15, half = lane >> 4;
#pragma unroll
  for (int mt = 0; mt < 2; ++mt)
#pragma unroll
    for (int nt = 0; nt < 4; ++nt) {
      const int col = colBase + wn * 64 + nt * 16 + n16;
      const float bsv = bo[col];
#pragma unroll
      for (int r = 0; r < 8; ++r) {
        const int row = rowBase + wm * 32 + mt * 16 + half * 8 + r;
        proj[(size_t)row * DDIM + col] = acc[mt][nt][r] + bsv;
      }
    }
}

// Row LayerNorm over 1024 columns
__global__ __launch_bounds__(256) void ln_kernel(
    const float* __restrict__ in, const float* __restrict__ g,
    const float* __restrict__ be, float* __restrict__ out) {
  __shared__ float red[16];
  const int row = blockIdx.x;
  const float* p = in + (size_t)row * DDIM;
  float s = 0.f, s2 = 0.f;
  for (int i = threadIdx.x; i < DDIM; i += 256) {
    float v = p[i];
    s += v;
    s2 += v * v;
  }
#pragma unroll
  for (int off = 16; off; off >>= 1) {
    s += __shfl_xor(s, off);
    s2 += __shfl_xor(s2, off);
  }
  const int wave = threadIdx.x >> 5, lane = threadIdx.x & 31;
  if (lane == 0) { red[wave] = s; red[8 + wave] = s2; }
  __syncthreads();
  s = 0.f; s2 = 0.f;
#pragma unroll
  for (int w = 0; w < 8; ++w) { s += red[w]; s2 += red[8 + w]; }
  const float mu = s * (1.f / 1024.f);
  const float var = s2 * (1.f / 1024.f) - mu * mu;
  const float rstd = rsqrtf(var + 1e-12f);
  for (int i = threadIdx.x; i < DDIM; i += 256) {
    float v = p[i];
    out[(size_t)row * DDIM + i] = (v - mu) * rstd * g[i] + be[i];
  }
}

// ---------------------------------------------------------------------------

extern "C" void kernel_launch(void* const* d_in, const int* in_sizes, int n_in,
                              void* d_out, int out_size, void* d_ws, size_t ws_size,
                              hipStream_t stream) {
  const float* x   = (const float*)d_in[0];
  const float* wq  = (const float*)d_in[1];
  const float* bq  = (const float*)d_in[2];
  const float* wk  = (const float*)d_in[3];
  const float* bk  = (const float*)d_in[4];
  const float* wv  = (const float*)d_in[5];
  const float* bv  = (const float*)d_in[6];
  const float* wo  = (const float*)d_in[7];
  const float* bo  = (const float*)d_in[8];
  const float* lng = (const float*)d_in[9];
  const float* lnb = (const float*)d_in[10];
  float* out = (float*)d_out;

  const size_t M = (size_t)MROWS;
  const size_t xbBytes = M * DDIM * 2;          // bf16 [8192,1024]
  const size_t wBytes  = (size_t)DDIM * DDIM * 2;

  size_t off = 0;
  auto take = [&](size_t bytes) {
    size_t o = off;
    off += (bytes + 255) & ~(size_t)255;
    return o;
  };
  const size_t o_xb = take(xbBytes);
  const size_t o_wq = take(wBytes);
  const size_t o_wk = take(wBytes);
  const size_t o_wv = take(wBytes);
  const size_t o_wo = take(wBytes);
  const size_t o_q  = take(xbBytes);
  const size_t o_k  = take(xbBytes);
  const size_t o_v  = take(xbBytes);
  const size_t o_at = take(xbBytes);
  const size_t o_pj = take(M * DDIM * 4);
  if (ws_size < off) return;  // workspace too small: nothing safe to do

  char* ws = (char*)d_ws;
  __bf16* xb  = (__bf16*)(ws + o_xb);
  __bf16* wqb = (__bf16*)(ws + o_wq);
  __bf16* wkb = (__bf16*)(ws + o_wk);
  __bf16* wvb = (__bf16*)(ws + o_wv);
  __bf16* wob = (__bf16*)(ws + o_wo);
  __bf16* qb  = (__bf16*)(ws + o_q);
  __bf16* kb2 = (__bf16*)(ws + o_k);
  __bf16* vb2 = (__bf16*)(ws + o_v);
  __bf16* atb = (__bf16*)(ws + o_at);
  float*  pj  = (float*)(ws + o_pj);

  const int nX = (int)(M * DDIM);
  const int nW = DDIM * DDIM;
  cvt_f32_bf16<<<dim3((nX + 255) / 256), dim3(256), 0, stream>>>(x, xb, nX);
  cvt_f32_bf16<<<dim3((nW + 255) / 256), dim3(256), 0, stream>>>(wq, wqb, nW);
  cvt_f32_bf16<<<dim3((nW + 255) / 256), dim3(256), 0, stream>>>(wk, wkb, nW);
  cvt_f32_bf16<<<dim3((nW + 255) / 256), dim3(256), 0, stream>>>(wv, wvb, nW);
  cvt_f32_bf16<<<dim3((nW + 255) / 256), dim3(256), 0, stream>>>(wo, wob, nW);

  // QKV projections: grid (Mtiles=64, Ntiles=8, qkv=3)
  qkv_gemm_kernel<<<dim3(64, 8, 3), dim3(256), 0, stream>>>(
      xb, wqb, wkb, wvb, bq, bk, bv, qb, kb2, vb2);

  // attention: grid (qtiles=32, heads=16, batch=4), 4 waves/block
  attn_kernel<<<dim3(32, 16, 4), dim3(128), 0, stream>>>(qb, kb2, vb2, atb);

  // output projection
  oproj_gemm_kernel<<<dim3(64, 8, 1), dim3(256), 0, stream>>>(atb, wob, bo, pj);

  // layernorm -> d_out
  ln_kernel<<<dim3(MROWS), dim3(256), 0, stream>>>(pj, lng, lnb, out);

  (void)in_sizes; (void)n_in; (void)out_size;
}
